// ST_HGAT_10668698763495
// MI455X (gfx1250) — matrix operand order repaired and checked by reading, not compiled, not run on hardware
//
#include <hip/hip_runtime.h>
#include <hip/hip_bf16.h>
#include <math.h>

// ---------------- types / constants ----------------
typedef __attribute__((ext_vector_type(16))) __bf16 v16bf;
typedef __attribute__((ext_vector_type(8)))  __bf16 v8bf;
typedef __attribute__((ext_vector_type(8)))  float  v8f;

#define HIDDIM 128
#define NEDGE  200000

static const int SRCT[7] = {0,1,2,0,0,1,2};   // edge-type source node type
static const int DSTT[7] = {0,1,2,1,2,0,0};   // edge-type dest node type
static const int NOFF[3] = {0, 50000, 75000}; // row offset in packed X
static const int NCNT[3] = {50000, 25000, 25000};

// ---------------- device helpers ----------------
__device__ __forceinline__ __bf16 f2bf(float f) {
  unsigned u = __float_as_uint(f);
  unsigned r = u + 0x7FFFu + ((u >> 16) & 1u);   // round-to-nearest-even
  unsigned short h = (unsigned short)(r >> 16);
  return __builtin_bit_cast(__bf16, h);
}
__device__ __forceinline__ float eluf(float v) {
  return v > 0.f ? v : (__expf(v) - 1.f);
}
// order-preserving float <-> u32 map for atomic max over signed floats
__device__ __forceinline__ unsigned fmapu(float f) {
  unsigned u = __float_as_uint(f);
  return (u & 0x80000000u) ? ~u : (u | 0x80000000u);
}
__device__ __forceinline__ float funmapu(unsigned u) {
  return __uint_as_float((u & 0x80000000u) ? (u & 0x7FFFFFFFu) : ~u);
}

// ---------------- kernels ----------------

// x = elu(xin @ W[D,128] + b); dual-store f32 + bf16 mirror
__global__ __launch_bounds__(256) void proj_elu_kernel(
    const float* __restrict__ Xin, const float* __restrict__ W,
    const float* __restrict__ b, float* __restrict__ Xout,
    __bf16* __restrict__ XBout, int N, int D) {
  int idx = blockIdx.x * blockDim.x + threadIdx.x;
  if (idx >= N * HIDDIM) return;
  int n = idx >> 7, f = idx & 127;
  float acc = b[f];
  for (int j = 0; j < D; ++j) acc += Xin[n * D + j] * W[j * HIDDIM + f];
  float v = eluf(acc);
  Xout[idx] = v;
  XBout[idx] = f2bf(v);
}

// WT[n,k] = bf16(W[k,n])  (128x128)
__global__ __launch_bounds__(256) void w_transpose_bf16_kernel(
    const float* __restrict__ W, __bf16* __restrict__ WT) {
  int idx = blockIdx.x * blockDim.x + threadIdx.x;   // 0..16383
  int n = idx >> 7, k = idx & 127;
  WT[(size_t)n * HIDDIM + k] = f2bf(W[(size_t)k * HIDDIM + n]);
}

// Y[N,128] = XB[N,128] @ W + bias (W given transposed bf16: WT[N][K])
// block = 128 threads = 4 waves; each wave computes a 16x32 tile (2 WMMA cols)
__global__ __launch_bounds__(128) void wmma_gemm_bias_kernel(
    const __bf16* __restrict__ XB, const __bf16* __restrict__ WT,
    const float* __restrict__ bias, float* __restrict__ Y, int Nrows) {
  const int lane = threadIdx.x & 31;
  const int wave = threadIdx.x >> 5;
  const int row0 = blockIdx.x * 16;
  const int col0 = wave * 32;                   // 4 waves x 32 cols = 128
  const int hl  = lane >> 4;                    // half-wave: 0 or 1
  const int l16 = lane & 15;

  int rowA = row0 + l16;
  if (rowA >= Nrows) rowA = Nrows - 1;          // clamp (dup rows, guarded store)
  const __bf16* xrow = XB + (size_t)rowA * HIDDIM;
  const int colB0 = col0 + l16;                 // tile 0 column
  const int colB1 = colB0 + 16;                 // tile 1 column
  const __bf16* wc0 = WT + (size_t)colB0 * HIDDIM;
  const __bf16* wc1 = WT + (size_t)colB1 * HIDDIM;
  const int kbA = hl * 16;                      // A K base (contiguous, matches B)
  const int kbB = hl * 16;                      // B K base (contiguous, matches A)

  v8f c0 = {}, c1 = {};
  #pragma unroll
  for (int kk = 0; kk < HIDDIM; kk += 32) {
    // A fragment: element i -> K = kbA + i
    v8bf alo = *(const v8bf*)(xrow + kk + kbA);
    v8bf ahi = *(const v8bf*)(xrow + kk + kbA + 8);
    v16bf a = __builtin_shufflevector(alo, ahi,
        0, 1, 2, 3, 4, 5, 6, 7, 8, 9, 10, 11, 12, 13, 14, 15);
    // B fragment: element i -> K = kbB + i, contiguous along WT row
    v16bf b0 = *(const v16bf*)(wc0 + kk + kbB);
    v16bf b1 = *(const v16bf*)(wc1 + kk + kbB);
    c0 = __builtin_amdgcn_wmma_f32_16x16x32_bf16(
        false, a, false, b0, (short)0, c0, false, false);
    c1 = __builtin_amdgcn_wmma_f32_16x16x32_bf16(
        false, a, false, b1, (short)0, c1, false, false);
  }
  const float bv0 = bias[colB0];
  const float bv1 = bias[colB1];
  #pragma unroll
  for (int r = 0; r < 8; ++r) {
    int row = row0 + r + hl * 8;                // C/D layout: M = r + half*8
    if (row < Nrows) {
      Y[(size_t)row * HIDDIM + colB0] = c0[r] + bv0;
      Y[(size_t)row * HIDDIM + colB1] = c1[r] + bv1;
    }
  }
}

// fill u32 buffer with pattern (for amax init = map(-inf))
__global__ void fill_u32_kernel(unsigned* p, unsigned v, int n) {
  int i = blockIdx.x * blockDim.x + threadIdx.x;
  if (i < n) p[i] = v;
}

// one wave per edge: alpha[e,h] = sum_c lrelu(xl[s]+xr[d]+ea@We)*att ; atomic max
__global__ __launch_bounds__(256) void edge_alpha_kernel(
    const float* __restrict__ XL, const float* __restrict__ XR,
    const float* __restrict__ EA, const float* __restrict__ We,
    const float* __restrict__ att, const int* __restrict__ si,
    const int* __restrict__ di, float* __restrict__ alpha,
    unsigned* __restrict__ amax, int E) {
  int gw = (blockIdx.x * blockDim.x + threadIdx.x) >> 5;
  int lane = threadIdx.x & 31;
  if (gw >= E) return;
  int s = si[gw], d = di[gw];
  const float4 xl4 = *(const float4*)(XL + (size_t)s * HIDDIM + lane * 4);
  const float4 xr4 = *(const float4*)(XR + (size_t)d * HIDDIM + lane * 4);
  float e0 = EA[(size_t)gw * 4 + 0], e1 = EA[(size_t)gw * 4 + 1];
  float e2 = EA[(size_t)gw * 4 + 2], e3 = EA[(size_t)gw * 4 + 3];
  const float* xlp = (const float*)&xl4;
  const float* xrp = (const float*)&xr4;
  float acc = 0.f;
  #pragma unroll
  for (int i = 0; i < 4; ++i) {
    int f = lane * 4 + i;                        // f = h*32 + c
    float m = xlp[i] + xrp[i] + e0 * We[f] + e1 * We[128 + f] +
              e2 * We[256 + f] + e3 * We[384 + f];
    float lr = m > 0.f ? m : 0.2f * m;           // leaky_relu 0.2
    acc += lr * att[f];                          // att flattened [H*C]
  }
  // reduce the 8 lanes belonging to each head (lanes 8h..8h+7)
  acc += __shfl_xor(acc, 1, 32);
  acc += __shfl_xor(acc, 2, 32);
  acc += __shfl_xor(acc, 4, 32);
  int h = lane >> 3;
  if ((lane & 7) == 0) {
    alpha[(size_t)gw * 4 + h] = acc;
    atomicMax(&amax[(size_t)d * 4 + h], fmapu(acc));
  }
}

// ex = exp(alpha - fixed(amax)); den[d,h] += ex   (thread per (edge,head))
__global__ __launch_bounds__(256) void edge_exp_kernel(
    const int* __restrict__ di, float* __restrict__ alpha,
    const unsigned* __restrict__ amax, float* __restrict__ den, int E) {
  int idx = blockIdx.x * blockDim.x + threadIdx.x;
  if (idx >= E * 4) return;
  int e = idx >> 2, h = idx & 3;
  int d = di[e];
  float am = funmapu(amax[(size_t)d * 4 + h]);
  if (am < -3.0e38f) am = 0.f;                   // where(isfinite, amax, 0)
  float ex = __expf(alpha[idx] - am);
  alpha[idx] = ex;                               // reuse buffer for ex
  atomicAdd(&den[(size_t)d * 4 + h], ex);
}

// one wave per edge: agg[d] += (ex/den) * xl[s]
__global__ __launch_bounds__(256) void edge_agg_kernel(
    const float* __restrict__ XL, const float* __restrict__ EXA,
    const float* __restrict__ den, const int* __restrict__ si,
    const int* __restrict__ di, float* __restrict__ AGGd, int E) {
  int gw = (blockIdx.x * blockDim.x + threadIdx.x) >> 5;
  int lane = threadIdx.x & 31;
  if (gw >= E) return;
  int s = si[gw], d = di[gw];
  int h = lane >> 3;
  float w = EXA[(size_t)gw * 4 + h] / (den[(size_t)d * 4 + h] + 1e-16f);
  const float4 xl4 = *(const float4*)(XL + (size_t)s * HIDDIM + lane * 4);
  float* dst = AGGd + (size_t)d * HIDDIM + lane * 4;
  atomicAdd(dst + 0, w * xl4.x);
  atomicAdd(dst + 1, w * xl4.y);
  atomicAdd(dst + 2, w * xl4.z);
  atomicAdd(dst + 3, w * xl4.w);
}

__global__ void bias_accum_kernel(float* bsum, const float* bias) {
  bsum[threadIdx.x] += bias[threadIdx.x];
}

// per-column sum/sumsq of y = elu(agg + bsum)
#define BN_ROWS 512
__global__ __launch_bounds__(256) void bn_stats_kernel(
    const float* __restrict__ A, const float* __restrict__ bsum,
    float* __restrict__ sums, float* __restrict__ sumsq, int N) {
  int f = threadIdx.x & 127;
  int rg = threadIdx.x >> 7;
  int base = blockIdx.x * BN_ROWS;
  int end = base + BN_ROWS; if (end > N) end = N;
  float bs = bsum[f];
  float s = 0.f, s2 = 0.f;
  for (int n = base + rg; n < end; n += 2) {
    float v = eluf(A[(size_t)n * HIDDIM + f] + bs);
    s += v; s2 += v * v;
  }
  atomicAdd(&sums[f], s);
  atomicAdd(&sumsq[f], s2);
}

// x = bn(elu(agg + bsum)); dual-store f32 + bf16 mirror
__global__ __launch_bounds__(256) void bn_norm_kernel(
    const float* __restrict__ A, const float* __restrict__ bsum,
    const float* __restrict__ sums, const float* __restrict__ sumsq,
    const float* __restrict__ gamma, const float* __restrict__ beta,
    float* __restrict__ Xo, __bf16* __restrict__ XBo, int N) {
  int idx = blockIdx.x * blockDim.x + threadIdx.x;
  if (idx >= N * HIDDIM) return;
  int f = idx & 127;
  float invN = 1.0f / (float)N;
  float mu = sums[f] * invN;
  float var = sumsq[f] * invN - mu * mu;
  float v = eluf(A[idx] + bsum[f]);
  float o = (v - mu) * rsqrtf(var + 1e-5f) * gamma[f] + beta[f];
  Xo[idx] = o;
  XBo[idx] = f2bf(o);
}

// heads
__global__ __launch_bounds__(256) void head_lidar_kernel(
    const float* __restrict__ X, const float* __restrict__ W,
    const float* __restrict__ b, float* __restrict__ out2,
    float* __restrict__ outs, int N) {
  int n = blockIdx.x * blockDim.x + threadIdx.x;
  if (n >= N) return;
  const float* xr = X + (size_t)n * HIDDIM;
  float o0 = b[0], o1 = b[1], o2 = b[2];
  for (int j = 0; j < HIDDIM; ++j) {
    float v = xr[j];
    o0 += v * W[j * 3 + 0]; o1 += v * W[j * 3 + 1]; o2 += v * W[j * 3 + 2];
  }
  out2[(size_t)n * 2 + 0] = o0;
  out2[(size_t)n * 2 + 1] = o1;
  outs[n] = 0.195f / (1.f + __expf(-o2)) + 0.005f;
}

__global__ __launch_bounds__(256) void head_radar_kernel(
    const float* __restrict__ X, const float* __restrict__ W,
    const float* __restrict__ b, float* __restrict__ outr,
    float* __restrict__ outs, int N) {
  int n = blockIdx.x * blockDim.x + threadIdx.x;
  if (n >= N) return;
  const float* xr = X + (size_t)n * HIDDIM;
  float o0 = b[0], o1 = b[1];
  for (int j = 0; j < HIDDIM; ++j) {
    float v = xr[j];
    o0 += v * W[j * 2 + 0]; o1 += v * W[j * 2 + 1];
  }
  outr[n] = o0;
  outs[n] = 0.97f / (1.f + __expf(-o1)) + 0.03f;
}

// ---------------- host launcher ----------------
extern "C" void kernel_launch(void* const* d_in, const int* in_sizes, int n_in,
                              void* d_out, int out_size, void* d_ws, size_t ws_size,
                              hipStream_t stream) {
  (void)in_sizes; (void)n_in; (void)out_size; (void)ws_size;
  // ---- input index map (setup_inputs insertion order) ----
  // 0..2: x_lidar, x_radar1, x_radar2
  // 3..9: edge_attrs[k]
  // 10..23: edge_indices: (src,dst) per k
  // 24..29: proj (W,b) x3
  // 30 + li*49 + k*7 + {0:Wl,1:bl,2:Wr,3:br,4:We,5:att,6:bias}
  // 128..139: bns (gamma,beta) per layer per nt
  // 140,141: head_lidar W,b ; 142,143: head_radar W,b
  auto fin = [&](int i) { return (const float*)d_in[i]; };
  auto iin = [&](int i) { return (const int*)d_in[i]; };

  // ---- workspace carve-up ----
  float* ws   = (float*)d_ws;
  float* X    = ws;                       // 100000*128 f32
  float* AGG  = X   + 100000 * HIDDIM;    // 100000*128 f32
  float* XL   = AGG + 100000 * HIDDIM;    // 50000*128 f32
  float* XR   = XL  + 50000 * HIDDIM;     // 50000*128 f32
  float* ALPHA= XR  + 50000 * HIDDIM;     // 200000*4 f32
  float* DEN  = ALPHA + NEDGE * 4;        // 200000 f32 (max Nd*4)
  unsigned* AMAX = (unsigned*)(DEN + 200000); // 200000 u32
  float* BSUM = (float*)(AMAX + 200000);  // 3*128
  float* BSS  = BSUM + 384;               // 3*128 col sums
  float* BSQ  = BSS + 384;                // 3*128 col sumsq
  __bf16* XB  = (__bf16*)(BSQ + 384);     // 100000*128 bf16 mirror of X
  __bf16* WT  = XB + (size_t)100000 * HIDDIM; // 128*128 bf16 transposed weight

  const unsigned MAP_NEG_INF = 0x007FFFFFu;  // fmapu(-inf)

  // ---- input projection: x = elu(xin @ Wp + b) ----
  proj_elu_kernel<<<(50000 * HIDDIM + 255) / 256, 256, 0, stream>>>(
      fin(0), fin(24), fin(25), X + (size_t)NOFF[0] * HIDDIM,
      XB + (size_t)NOFF[0] * HIDDIM, 50000, 6);
  proj_elu_kernel<<<(25000 * HIDDIM + 255) / 256, 256, 0, stream>>>(
      fin(1), fin(26), fin(27), X + (size_t)NOFF[1] * HIDDIM,
      XB + (size_t)NOFF[1] * HIDDIM, 25000, 5);
  proj_elu_kernel<<<(25000 * HIDDIM + 255) / 256, 256, 0, stream>>>(
      fin(2), fin(28), fin(29), X + (size_t)NOFF[2] * HIDDIM,
      XB + (size_t)NOFF[2] * HIDDIM, 25000, 5);

  // ---- GNN layers ----
  for (int li = 0; li < 2; ++li) {
    hipMemsetAsync(AGG, 0, (size_t)100000 * HIDDIM * sizeof(float), stream);
    hipMemsetAsync(BSUM, 0, 384 * sizeof(float), stream);

    for (int k = 0; k < 7; ++k) {
      const int st = SRCT[k], dt = DSTT[k];
      const int Ns = NCNT[st], Nd = NCNT[dt];
      const int cb = 30 + li * 49 + k * 7;
      const float* Wl = fin(cb + 0); const float* bl = fin(cb + 1);
      const float* Wr = fin(cb + 2); const float* br = fin(cb + 3);
      const float* We = fin(cb + 4); const float* att = fin(cb + 5);
      const float* cbias = fin(cb + 6);
      const float* EA = fin(3 + k);
      const int* si = iin(10 + 2 * k);
      const int* di = iin(11 + 2 * k);

      // xl = x_src @ Wl + bl  (transpose weight -> bf16, then WMMA)
      w_transpose_bf16_kernel<<<64, 256, 0, stream>>>(Wl, WT);
      wmma_gemm_bias_kernel<<<(Ns + 15) / 16, 128, 0, stream>>>(
          XB + (size_t)NOFF[st] * HIDDIM, WT, bl, XL, Ns);
      // xr = x_dst @ Wr + br
      w_transpose_bf16_kernel<<<64, 256, 0, stream>>>(Wr, WT);
      wmma_gemm_bias_kernel<<<(Nd + 15) / 16, 128, 0, stream>>>(
          XB + (size_t)NOFF[dt] * HIDDIM, WT, br, XR, Nd);

      // segment softmax scratch init
      fill_u32_kernel<<<(Nd * 4 + 255) / 256, 256, 0, stream>>>(AMAX, MAP_NEG_INF, Nd * 4);
      hipMemsetAsync(DEN, 0, (size_t)Nd * 4 * sizeof(float), stream);

      // edge passes
      edge_alpha_kernel<<<(NEDGE * 32 + 255) / 256, 256, 0, stream>>>(
          XL, XR, EA, We, att, si, di, ALPHA, AMAX, NEDGE);
      edge_exp_kernel<<<(NEDGE * 4 + 255) / 256, 256, 0, stream>>>(
          di, ALPHA, AMAX, DEN, NEDGE);
      edge_agg_kernel<<<(NEDGE * 32 + 255) / 256, 256, 0, stream>>>(
          XL, ALPHA, DEN, si, di, AGG + (size_t)NOFF[dt] * HIDDIM, NEDGE);

      // conv bias is added for every dst node regardless of edges
      bias_accum_kernel<<<1, 128, 0, stream>>>(BSUM + dt * HIDDIM, cbias);
    }

    // batch norm over y = elu(agg + bsum) per node type
    hipMemsetAsync(BSS, 0, 384 * sizeof(float), stream);
    hipMemsetAsync(BSQ, 0, 384 * sizeof(float), stream);
    for (int nt = 0; nt < 3; ++nt) {
      const int N = NCNT[nt];
      const float* gamma = fin(128 + li * 6 + nt * 2 + 0);
      const float* beta  = fin(128 + li * 6 + nt * 2 + 1);
      float* Ap = AGG + (size_t)NOFF[nt] * HIDDIM;
      bn_stats_kernel<<<(N + BN_ROWS - 1) / BN_ROWS, 256, 0, stream>>>(
          Ap, BSUM + nt * HIDDIM, BSS + nt * HIDDIM, BSQ + nt * HIDDIM, N);
      bn_norm_kernel<<<(N * HIDDIM + 255) / 256, 256, 0, stream>>>(
          Ap, BSUM + nt * HIDDIM, BSS + nt * HIDDIM, BSQ + nt * HIDDIM,
          gamma, beta, X + (size_t)NOFF[nt] * HIDDIM,
          XB + (size_t)NOFF[nt] * HIDDIM, N);
    }
  }

  // ---- heads -> d_out (concatenated in reference return order) ----
  float* out = (float*)d_out;
  float* O0 = out;            // lidar xy      [50000,2]
  float* O1 = out + 100000;   // lidar sigma   [50000]
  float* O2 = out + 150000;   // radar1 v      [25000]
  float* O3 = out + 175000;   // radar1 sigma  [25000]
  float* O4 = out + 200000;   // radar2 v      [25000]
  float* O5 = out + 225000;   // radar2 sigma  [25000]

  head_lidar_kernel<<<(50000 + 255) / 256, 256, 0, stream>>>(
      X + (size_t)NOFF[0] * HIDDIM, fin(140), fin(141), O0, O1, 50000);
  head_radar_kernel<<<(25000 + 255) / 256, 256, 0, stream>>>(
      X + (size_t)NOFF[1] * HIDDIM, fin(142), fin(143), O2, O3, 25000);
  head_radar_kernel<<<(25000 + 255) / 256, 256, 0, stream>>>(
      X + (size_t)NOFF[2] * HIDDIM, fin(142), fin(143), O4, O5, 25000);
}
